// VQVAE_56315611185435
// MI455X (gfx1250) — matrix operand order repaired and checked by reading, not compiled, not run on hardware
//
#include <hip/hip_runtime.h>
#include <hip/hip_bf16.h>

typedef __attribute__((ext_vector_type(16))) _Float16 v16h;
typedef __attribute__((ext_vector_type(8)))  _Float16 v8h;
typedef __attribute__((ext_vector_type(8)))  float    v8f;

namespace {
constexpr int NROWS  = 4 * 181 * 360;   // 260640, divisible by 32
constexpr int MTILE  = 32;              // rows per wave-tile (2 x 16-row WMMA subtiles)
constexpr int NTILES = NROWS / MTILE;   // 8145
constexpr int BLOCK  = 256;
constexpr int WPB    = BLOCK / 32;      // 8 waves per block (wave32)
constexpr int GRID   = 512;
constexpr int SLOTS  = GRID * WPB;      // 4096 loss partials
constexpr int ACT_STRIDE = 96;          // halves per activation row (max width 96)

// ---- LDS layout (byte offsets), all 16B aligned ----
constexpr int OFF_EW0  = 0;                       // [64][96] f16  (enc 89->64, W^T, K pad 96)
constexpr int OFF_EW1  = OFF_EW0 + 64*96*2;       // [32][64]
constexpr int OFF_EW2  = OFF_EW1 + 32*64*2;       // [32][32]  (N real 24)
constexpr int OFF_EW3  = OFF_EW2 + 32*32*2;       // [16][32]  (K real 24)
constexpr int OFF_CB   = OFF_EW3 + 16*32*2;       // [1024][16] codebook f16 (K real 16)
constexpr int OFF_DW0  = OFF_CB  + 1024*16*2;     // [32][16]  (dec 16->24)
constexpr int OFF_DW1  = OFF_DW0 + 32*16*2;       // [32][32]  (K real 24)
constexpr int OFF_DW2  = OFF_DW1 + 32*32*2;       // [64][32]
constexpr int OFF_DW3  = OFF_DW2 + 64*32*2;       // [96][64]  (N real 89)
constexpr int OFF_BIAS = OFF_DW3 + 96*64*2;       // 368 f32 (padded biases)
constexpr int OFF_CNORM= OFF_BIAS + 368*4;        // 1024 f32: -0.5*|c|^2
constexpr int OFF_IDX  = OFF_CNORM + 1024*4;      // WPB*32 ints (argmin per row)
constexpr int OFF_ZERO = OFF_IDX + WPB*32*4;      // 32B of f16 zeros (branchless K16 B)
constexpr int OFF_ACT  = OFF_ZERO + 32;           // per-wave [32][96] f16
constexpr int LDS_BYTES= OFF_ACT + WPB*MTILE*ACT_STRIDE*2;  // = 127456 B

// bias sub-offsets (floats)
constexpr int B_E0 = 0, B_E1 = 64, B_E2 = 96, B_E3 = 128;
constexpr int B_D0 = 144, B_D1 = 176, B_D2 = 208, B_D3 = 272;
}

union HV { v16h v; v8h h[2]; };

__device__ __forceinline__ v8h zero8() {
  v8h z = { (_Float16)0, (_Float16)0, (_Float16)0, (_Float16)0,
            (_Float16)0, (_Float16)0, (_Float16)0, (_Float16)0 };
  return z;
}
__device__ __forceinline__ v8f zero8f() {
  v8f z = {0.f,0.f,0.f,0.f,0.f,0.f,0.f,0.f};
  return z;
}

// A fragment (16x32 f16): lane<16 holds row m=lane, K {kt*32+0..7, +16..23};
// lane>=16 holds row m=lane-16, K {kt*32+8..15, +24..31}. Two ds_load_b128.
__device__ __forceinline__ v16h load_a_frag(const _Float16* act, int lane, int kt) {
  const int m   = lane & 15;
  const int off = (lane & 16) ? 8 : 0;
  const _Float16* p = act + m * ACT_STRIDE + kt * 32 + off;
  HV u;
  u.h[0] = *(const v8h*)p;
  u.h[1] = *(const v8h*)(p + 16);
  return u.v;
}

// A fragment with only K=0..15 real (z / z_q): upper K-half is zero.
__device__ __forceinline__ v16h load_a_frag_k16(const _Float16* act, int lane) {
  const int m   = lane & 15;
  const int off = (lane & 16) ? 8 : 0;
  HV u;
  u.h[0] = *(const v8h*)(act + m * ACT_STRIDE + off);
  u.h[1] = zero8();
  return u.v;
}

// B fragment (32x16 f16) from W^T[N][KPAD] in LDS: column n=lane&15 (+16*nt),
// lanes 0-15 K={kt*32+0..7, +16..23}, lanes 16-31 K={kt*32+8..15, +24..31}.
template<int KPAD>
__device__ __forceinline__ v16h load_b_frag(const _Float16* Wt, int lane, int kt, int nt) {
  const int n  = (lane & 15) + nt * 16;
  const int kb = kt * 32 + ((lane & 16) ? 8 : 0);
  const _Float16* p = Wt + n * KPAD + kb;
  HV u;
  u.h[0] = *(const v8h*)p;
  u.h[1] = *(const v8h*)(p + 16);
  return u.v;
}

// B fragment, K real 16, stored [N][16]: lanes<16 K=0..7, lanes>=16 K=8..15;
// upper half of every lane reads the zero region (no EXEC manipulation).
__device__ __forceinline__ v16h load_b_frag_k16(const _Float16* Wt, const _Float16* zreg,
                                                int lane, int nt) {
  const _Float16* p = Wt + ((lane & 15) + nt * 16) * 16 + ((lane & 16) ? 8 : 0);
  HV u;
  u.h[0] = *(const v8h*)p;
  u.h[1] = *(const v8h*)zreg;
  return u.v;
}

__device__ __forceinline__ v8f wmma32f16(v16h a, v16h b, v8f c) {
  return __builtin_amdgcn_wmma_f32_16x16x32_f16(false, a, false, b, (short)0, c, false, false);
}

__device__ __forceinline__ void store_sub(_Float16* act, int s, int lane,
                                          const v8f* acc, const float* bias, int NT, bool relu) {
  const int mo = (lane & 16) ? 8 : 0;
  const int nl = lane & 15;
  for (int nt = 0; nt < NT; ++nt) {
    const float b = bias[nt * 16 + nl];
    #pragma unroll
    for (int v = 0; v < 8; ++v) {
      float val = acc[nt][v] + b;
      if (relu) val = fmaxf(val, 0.0f);
      act[(s * 16 + v + mo) * ACT_STRIDE + nt * 16 + nl] = (_Float16)val;
    }
  }
}

// In-place dense layer on a per-wave [32][96] f16 tile: two 16-row subtiles
// share every B fragment (2 WMMAs per B load). All input reads precede stores.
template<int KT, int NT>
__device__ __forceinline__ void dense_layer2(_Float16* act, const _Float16* Wt,
                                             const float* bias, int lane, bool relu) {
  v8f acc0[NT], acc1[NT];
  #pragma unroll
  for (int nt = 0; nt < NT; ++nt) { acc0[nt] = zero8f(); acc1[nt] = zero8f(); }
  #pragma unroll
  for (int kt = 0; kt < KT; ++kt) {
    const v16h A0 = load_a_frag(act, lane, kt);
    const v16h A1 = load_a_frag(act + 16 * ACT_STRIDE, lane, kt);
    #pragma unroll
    for (int nt = 0; nt < NT; ++nt) {
      const v16h B = load_b_frag<KT * 32>(Wt, lane, kt, nt);
      acc0[nt] = wmma32f16(A0, B, acc0[nt]);
      acc1[nt] = wmma32f16(A1, B, acc1[nt]);
    }
  }
  __builtin_amdgcn_wave_barrier();
  store_sub(act, 0, lane, acc0, bias, NT, relu);
  store_sub(act, 1, lane, acc1, bias, NT, relu);
  __builtin_amdgcn_wave_barrier();
}

// Dense layer whose input has K=16 real (decoder first layer on z_q).
template<int NT>
__device__ __forceinline__ void dense_layer2_k16(_Float16* act, const _Float16* Wt,
                                                 const _Float16* zreg, const float* bias,
                                                 int lane, bool relu) {
  const v16h A0 = load_a_frag_k16(act, lane);
  const v16h A1 = load_a_frag_k16(act + 16 * ACT_STRIDE, lane);
  v8f acc0[NT], acc1[NT];
  #pragma unroll
  for (int nt = 0; nt < NT; ++nt) {
    const v16h B = load_b_frag_k16(Wt, zreg, lane, nt);
    acc0[nt] = wmma32f16(A0, B, zero8f());
    acc1[nt] = wmma32f16(A1, B, zero8f());
  }
  __builtin_amdgcn_wave_barrier();
  store_sub(act, 0, lane, acc0, bias, NT, relu);
  store_sub(act, 1, lane, acc1, bias, NT, relu);
  __builtin_amdgcn_wave_barrier();
}

// Final 64->89 layer, N-tiles [ntBase, ntBase+NT): fp32 accumulate, store to global.
template<int KT, int NT>
__device__ __forceinline__ void dense_out2(const _Float16* act, const _Float16* Wt,
                                           const float* bias, int lane, int ntBase,
                                           float* __restrict__ orow) {
  v8f acc0[NT], acc1[NT];
  #pragma unroll
  for (int nt = 0; nt < NT; ++nt) { acc0[nt] = zero8f(); acc1[nt] = zero8f(); }
  #pragma unroll
  for (int kt = 0; kt < KT; ++kt) {
    const v16h A0 = load_a_frag(act, lane, kt);
    const v16h A1 = load_a_frag(act + 16 * ACT_STRIDE, lane, kt);
    #pragma unroll
    for (int nt = 0; nt < NT; ++nt) {
      const v16h B = load_b_frag<KT * 32>(Wt, lane, kt, ntBase + nt);
      acc0[nt] = wmma32f16(A0, B, acc0[nt]);
      acc1[nt] = wmma32f16(A1, B, acc1[nt]);
    }
  }
  const int mo = (lane & 16) ? 8 : 0;
  const int nl = lane & 15;
  #pragma unroll
  for (int nt = 0; nt < NT; ++nt) {
    const int n = (ntBase + nt) * 16 + nl;
    if (n < 89) {
      const float b = bias[n];
      #pragma unroll
      for (int v = 0; v < 8; ++v) {
        orow[(size_t)(v + mo) * 89 + n]        = acc0[nt][v] + b;
        orow[(size_t)(16 + v + mo) * 89 + n]   = acc1[nt][v] + b;
      }
    }
  }
}

// Stage W[K][N] (f32, global) -> W^T[NPAD][KPAD] (f16, LDS), zero padded.
__device__ __forceinline__ void stage_w(_Float16* dst, const float* W,
                                        int K, int N, int KPAD, int NPAD, int tid) {
  const int tot = NPAD * KPAD;
  for (int i = tid; i < tot; i += BLOCK) {
    const int n = i / KPAD;
    const int k = i - n * KPAD;
    const float v = (n < N && k < K) ? W[k * N + n] : 0.0f;
    dst[i] = (_Float16)v;
  }
}

__device__ __forceinline__ void stage_b(float* dst, const float* b, int N, int NPAD, int tid) {
  for (int i = tid; i < NPAD; i += BLOCK) dst[i] = (i < N) ? b[i] : 0.0f;
}

__global__ __launch_bounds__(BLOCK) void vqvae_fused(
    const float* __restrict__ x,
    const float* __restrict__ eW0, const float* __restrict__ eb0,
    const float* __restrict__ eW1, const float* __restrict__ eb1,
    const float* __restrict__ eW2, const float* __restrict__ eb2,
    const float* __restrict__ eW3, const float* __restrict__ eb3,
    const float* __restrict__ cb,
    const float* __restrict__ dW0, const float* __restrict__ db0,
    const float* __restrict__ dW1, const float* __restrict__ db1,
    const float* __restrict__ dW2, const float* __restrict__ db2,
    const float* __restrict__ dW3, const float* __restrict__ db3,
    float* __restrict__ out, float* __restrict__ partials)
{
  extern __shared__ __align__(16) char smem[];
  const int tid = threadIdx.x;

  // ---- one-time per-workgroup weight staging into LDS (f16, transposed) ----
  stage_w((_Float16*)(smem + OFF_EW0), eW0, 89, 64, 96, 64, tid);
  stage_w((_Float16*)(smem + OFF_EW1), eW1, 64, 32, 64, 32, tid);
  stage_w((_Float16*)(smem + OFF_EW2), eW2, 32, 24, 32, 32, tid);
  stage_w((_Float16*)(smem + OFF_EW3), eW3, 24, 16, 32, 16, tid);
  stage_w((_Float16*)(smem + OFF_DW0), dW0, 16, 24, 16, 32, tid);
  stage_w((_Float16*)(smem + OFF_DW1), dW1, 24, 32, 32, 32, tid);
  stage_w((_Float16*)(smem + OFF_DW2), dW2, 32, 64, 32, 64, tid);
  stage_w((_Float16*)(smem + OFF_DW3), dW3, 64, 89, 64, 96, tid);
  { // codebook [1024][16] row-major == desired B^T layout (K real 16)
    _Float16* d = (_Float16*)(smem + OFF_CB);
    for (int i = tid; i < 1024 * 16; i += BLOCK) d[i] = (_Float16)cb[i];
  }
  { // -0.5*|c|^2 bias for argmax-score form of the distance
    float* cn = (float*)(smem + OFF_CNORM);
    for (int n = tid; n < 1024; n += BLOCK) {
      float s = 0.f;
      #pragma unroll
      for (int k = 0; k < 16; ++k) { const float v = cb[n * 16 + k]; s += v * v; }
      cn[n] = -0.5f * s;
    }
  }
  if (tid < 16) ((_Float16*)(smem + OFF_ZERO))[tid] = (_Float16)0;
  float* bs = (float*)(smem + OFF_BIAS);
  stage_b(bs + B_E0, eb0, 64, 64, tid);
  stage_b(bs + B_E1, eb1, 32, 32, tid);
  stage_b(bs + B_E2, eb2, 24, 32, tid);
  stage_b(bs + B_E3, eb3, 16, 16, tid);
  stage_b(bs + B_D0, db0, 24, 32, tid);
  stage_b(bs + B_D1, db1, 32, 32, tid);
  stage_b(bs + B_D2, db2, 64, 64, tid);
  stage_b(bs + B_D3, db3, 89, 96, tid);
  __syncthreads();

  const int lane = tid & 31;
  const int wave = tid >> 5;
  _Float16* act = (_Float16*)(smem + OFF_ACT) + wave * MTILE * ACT_STRIDE;
  int* idxArr   = (int*)(smem + OFF_IDX) + wave * MTILE;
  const float* cnorm     = (const float*)(smem + OFF_CNORM);
  const _Float16* CB     = (const _Float16*)(smem + OFF_CB);
  const _Float16* zreg   = (const _Float16*)(smem + OFF_ZERO);

  float lloss = 0.0f;

  for (int tile = blockIdx.x * WPB + wave; tile < NTILES; tile += gridDim.x * WPB) {
    const int r0 = tile * MTILE;

    // ---- prefetch next grid-stride tile of x (gfx1250 global_prefetch_b8) ----
    {
      const int nxt = tile + gridDim.x * WPB;
      if (nxt < NTILES) {
        const char* pf = (const char*)(x + (size_t)nxt * MTILE * 89);
        __builtin_prefetch(pf + (size_t)lane * 128, 0, 1);
        __builtin_prefetch(pf + 4096 + (size_t)lane * 128, 0, 1);
        __builtin_prefetch(pf + 8192 + (size_t)lane * 128, 0, 1);
      }
    }

    // ---- load x tile: [32][89] f32 -> [32][96] f16, 32 loads in flight/chunk ----
    {
      const float* xrow = x + (size_t)r0 * 89;
      #pragma unroll
      for (int c = 0; c < 3; ++c) {
        float t[32];
        #pragma unroll
        for (int j = 0; j < 32; ++j) {
          const int i = c * 1024 + j * 32 + lane;
          const int m = i / ACT_STRIDE;
          const int k = i - m * ACT_STRIDE;
          t[j] = (k < 89) ? xrow[m * 89 + k] : 0.0f;
        }
        #pragma unroll
        for (int j = 0; j < 32; ++j)
          act[c * 1024 + j * 32 + lane] = (_Float16)t[j];
      }
    }
    __builtin_amdgcn_wave_barrier();

    // ---- encoder ----
    dense_layer2<3, 4>(act, (const _Float16*)(smem + OFF_EW0), bs + B_E0, lane, true); // 89->64
    dense_layer2<2, 2>(act, (const _Float16*)(smem + OFF_EW1), bs + B_E1, lane, true); // 64->32
    dense_layer2<1, 2>(act, (const _Float16*)(smem + OFF_EW2), bs + B_E2, lane, true); // 32->24
    dense_layer2<1, 1>(act, (const _Float16*)(smem + OFF_EW3), bs + B_E3, lane, true); // 24->16

    // ---- VQ: argmin ||z-c||^2 == argmax (z.c - 0.5|c|^2); 128 WMMAs/tile,
    //      each codebook B fragment reused by both 16-row subtiles ----
    const v16h Az0 = load_a_frag_k16(act, lane);
    const v16h Az1 = load_a_frag_k16(act + 16 * ACT_STRIDE, lane);
    float bv0[8], bv1[8]; int bi0[8], bi1[8];
    #pragma unroll
    for (int v = 0; v < 8; ++v) { bv0[v] = -3.4e38f; bi0[v] = 0; bv1[v] = -3.4e38f; bi1[v] = 0; }
    #pragma unroll 2
    for (int nt = 0; nt < 64; ++nt) {
      const v16h B = load_b_frag_k16(CB, zreg, lane, nt);
      const v8f s0 = wmma32f16(Az0, B, zero8f());
      const v8f s1 = wmma32f16(Az1, B, zero8f());
      const int n = nt * 16 + (lane & 15);
      const float c = cnorm[n];
      #pragma unroll
      for (int v = 0; v < 8; ++v) {
        const float v0 = s0[v] + c;            // strictly-greater keeps first idx
        if (v0 > bv0[v]) { bv0[v] = v0; bi0[v] = n; }
        const float v1 = s1[v] + c;
        if (v1 > bv1[v]) { bv1[v] = v1; bi1[v] = n; }
      }
    }
    // reduce argmax across the 16 lanes of each half-wave
    #pragma unroll
    for (int d = 8; d > 0; d >>= 1) {
      #pragma unroll
      for (int v = 0; v < 8; ++v) {
        float ov = __shfl_xor(bv0[v], d, 32); int oi = __shfl_xor(bi0[v], d, 32);
        if (ov > bv0[v] || (ov == bv0[v] && oi < bi0[v])) { bv0[v] = ov; bi0[v] = oi; }
        ov = __shfl_xor(bv1[v], d, 32); oi = __shfl_xor(bi1[v], d, 32);
        if (ov > bv1[v] || (ov == bv1[v] && oi < bi1[v])) { bv1[v] = ov; bi1[v] = oi; }
      }
    }
    if ((lane & 15) == 0) {
      const int base = (lane & 16) ? 8 : 0;
      #pragma unroll
      for (int v = 0; v < 8; ++v) {
        idxArr[base + v]      = bi0[v];
        idxArr[16 + base + v] = bi1[v];
      }
    }
    __builtin_amdgcn_wave_barrier();

    // ---- gather z_q (fp32 global codebook), loss partial, overwrite z in act ----
    {
      const int ci = idxArr[lane];               // lane == row (0..31)
      const float* crow = cb + (size_t)ci * 16;
      float s = 0.f;
      #pragma unroll
      for (int j = 0; j < 16; ++j) {
        const float cv = crow[j];
        const float zv = (float)act[lane * ACT_STRIDE + j];
        const float df = cv - zv;
        s += df * df;
        act[lane * ACT_STRIDE + j] = (_Float16)cv;  // straight-through: z_q
      }
      lloss += s;
    }
    __builtin_amdgcn_wave_barrier();

    // ---- decoder ----
    dense_layer2_k16<2>(act, (const _Float16*)(smem + OFF_DW0), zreg, bs + B_D0, lane, true); // 16->24
    dense_layer2<1, 2>(act, (const _Float16*)(smem + OFF_DW1), bs + B_D1, lane, true);        // 24->32
    dense_layer2<1, 4>(act, (const _Float16*)(smem + OFF_DW2), bs + B_D2, lane, true);        // 32->64
    float* orow = out + (size_t)r0 * 89;                                                      // 64->89
    dense_out2<2, 3>(act, (const _Float16*)(smem + OFF_DW3), bs + B_D3, lane, 0, orow);
    dense_out2<2, 3>(act, (const _Float16*)(smem + OFF_DW3), bs + B_D3, lane, 3, orow);
  }

  // ---- deterministic loss partials: one fixed slot per wave ----
  #pragma unroll
  for (int d = 16; d > 0; d >>= 1) lloss += __shfl_xor(lloss, d, 32);
  if (lane == 0) partials[blockIdx.x * WPB + wave] = lloss;
}

__global__ void vqvae_finalize(const float* __restrict__ partials, float* __restrict__ out) {
  double s = 0.0;
  for (int i = 0; i < SLOTS; ++i) s += (double)partials[i];
  // vq_loss = mean + COMMIT*mean = 1.25 * sum / (N*16)
  out[(size_t)NROWS * 89] = (float)(1.25 * s / ((double)NROWS * 16.0));
}

extern "C" void kernel_launch(void* const* d_in, const int* in_sizes, int n_in,
                              void* d_out, int out_size, void* d_ws, size_t ws_size,
                              hipStream_t stream) {
  (void)in_sizes; (void)n_in; (void)out_size; (void)ws_size;
  // setup_inputs() dict order: x, (enc_W0,enc_b0)..(enc_W3,enc_b3), codebook,
  // (dec_W0,dec_b0)..(dec_W3,dec_b3)
  const float* x   = (const float*)d_in[0];
  const float* eW0 = (const float*)d_in[1];  const float* eb0 = (const float*)d_in[2];
  const float* eW1 = (const float*)d_in[3];  const float* eb1 = (const float*)d_in[4];
  const float* eW2 = (const float*)d_in[5];  const float* eb2 = (const float*)d_in[6];
  const float* eW3 = (const float*)d_in[7];  const float* eb3 = (const float*)d_in[8];
  const float* cb  = (const float*)d_in[9];
  const float* dW0 = (const float*)d_in[10]; const float* db0 = (const float*)d_in[11];
  const float* dW1 = (const float*)d_in[12]; const float* db1 = (const float*)d_in[13];
  const float* dW2 = (const float*)d_in[14]; const float* db2 = (const float*)d_in[15];
  const float* dW3 = (const float*)d_in[16]; const float* db3 = (const float*)d_in[17];
  float* out      = (float*)d_out;
  float* partials = (float*)d_ws;

  // ~124KB dynamic LDS: raise the per-kernel dynamic shared memory cap.
  (void)hipFuncSetAttribute((const void*)vqvae_fused,
                            hipFuncAttributeMaxDynamicSharedMemorySize, LDS_BYTES);

  vqvae_fused<<<GRID, BLOCK, LDS_BYTES, stream>>>(
      x, eW0, eb0, eW1, eb1, eW2, eb2, eW3, eb3, cb,
      dW0, db0, dW1, db1, dW2, db2, dW3, db3, out, partials);
  vqvae_finalize<<<1, 1, 0, stream>>>(partials, out);
}
